// Attention_18459769438488
// MI455X (gfx1250) — matrix operand order repaired and checked by reading, not compiled, not run on hardware
//
#include <hip/hip_runtime.h>
#include <hip/hip_bf16.h>

// ---------------------------------------------------------------------------
// MI455X (gfx1250) CoaT factorized-attention block.
// Problem: B=4, N=2304 (48x48), C=256, heads=8, ch=32.
// - All GEMM-shaped math on v_wmma_f32_16x16x32_bf16 (wave32).
// - Flash-style attention: no NxN score materialization; k/v tiles staged
//   into LDS once per block (4 waves share them) via async global->LDS DMA
//   (ASYNCcnt) with double buffering; s_wait_asynccnt + barrier for sync.
// - CRPE convs/lowpass on VALU with LDS plane tiling + global prefetch.
// ---------------------------------------------------------------------------

typedef __attribute__((ext_vector_type(16))) __bf16 bf16x16;
typedef __attribute__((ext_vector_type(8)))  __bf16 bf16x8;
typedef __attribute__((ext_vector_type(8)))  float  f32x8;
typedef int v4i_vec __attribute__((__vector_size__(16)));   // b128 payload type

#define NB    4
#define NTOK  2304      // 48*48
#define CDIM  256
#define NHEAD 8
#define CH    32
#define MROWS (NB*NTOK) // 9216
#define NJT   (NTOK/32) // 72 column tiles in attention

// ---------------- workspace layout (all sizes are multiples of 256B) -------
static constexpr size_t SZ_XB  = (size_t)MROWS*CDIM*2;     // x bf16
static constexpr size_t SZ_W1  = (size_t)1024*256*2;       // [wq;wkv;wres] bf16
static constexpr size_t SZ_WP  = (size_t)256*256*2;        // wproj bf16
static constexpr size_t SZ_QF  = (size_t)MROWS*CDIM*4;     // q f32 [b,h,n,ch]
static constexpr size_t SZ_QB  = SZ_XB;                    // q bf16 [b,h,n,ch]
static constexpr size_t SZ_KB  = SZ_XB;                    // k bf16 [b,h,n,ch]
static constexpr size_t SZ_VT  = SZ_XB;                    // v bf16 [b,h,ch,n]
static constexpr size_t SZ_RES = SZ_QF;                    // res f32 NCHW
static constexpr size_t SZ_K3  = (size_t)64*64*9*4;
static constexpr size_t SZ_K5  = (size_t)64*64*25*4;
static constexpr size_t SZ_K7  = (size_t)64*64*49*4;
static constexpr size_t SZ_CV  = (size_t)NB*192*NTOK*4;    // conv chunks 1..3
static constexpr size_t SZ_PL  = (size_t)12800*4;          // pooled grids
static constexpr size_t SZ_AT  = SZ_QF;                    // attn out f32 [b,n,c]
static constexpr size_t SZ_ZB  = SZ_XB;                    // (attn+crpe) bf16

static constexpr size_t OFF_XB  = 0;
static constexpr size_t OFF_W1  = OFF_XB  + SZ_XB;
static constexpr size_t OFF_WP  = OFF_W1  + SZ_W1;
static constexpr size_t OFF_QF  = OFF_WP  + SZ_WP;
static constexpr size_t OFF_QB  = OFF_QF  + SZ_QF;
static constexpr size_t OFF_KB  = OFF_QB  + SZ_QB;
static constexpr size_t OFF_VT  = OFF_KB  + SZ_KB;
static constexpr size_t OFF_RES = OFF_VT  + SZ_VT;
static constexpr size_t OFF_K3  = OFF_RES + SZ_RES;
static constexpr size_t OFF_K5  = OFF_K3  + SZ_K3;
static constexpr size_t OFF_K7  = OFF_K5  + SZ_K5;
static constexpr size_t OFF_CV  = OFF_K7  + SZ_K7;
static constexpr size_t OFF_PL  = OFF_CV  + SZ_CV;
static constexpr size_t OFF_AT  = OFF_PL  + SZ_PL;
static constexpr size_t OFF_ZB  = OFF_AT  + SZ_AT;

// ---------------- CDNA5 async global->LDS staging ---------------------------
#if defined(__gfx1250__) && __has_builtin(__builtin_amdgcn_global_load_async_to_lds_b128)
#define HAVE_ASYNC 1
#else
#define HAVE_ASYNC 0
#endif

// Copy 16B (8 x bf16) from global to LDS. Async path uses the gfx1250
// GLOBAL_LOAD_ASYNC_TO_LDS_B128 (ASYNCcnt); fallback is load + ds_store.
// Builtin signature (from hipcc diagnostic): (v4i AS1*, v4i AS3*, imm, imm).
__device__ __forceinline__ void stage16(const __bf16* g, __bf16* l) {
#if defined(__HIP_DEVICE_COMPILE__)
#if HAVE_ASYNC
  __builtin_amdgcn_global_load_async_to_lds_b128(
      (__attribute__((address_space(1))) v4i_vec*)g,
      (__attribute__((address_space(3))) v4i_vec*)l, 0, 0);
#else
  *(bf16x8*)l = *(const bf16x8*)g;
#endif
#endif
}

template <int N>
__device__ __forceinline__ void wait_asynccnt() {
#if defined(__HIP_DEVICE_COMPILE__)
#if __has_builtin(__builtin_amdgcn_s_wait_asynccnt)
  __builtin_amdgcn_s_wait_asynccnt(N);
#else
  asm volatile("s_wait_asynccnt %0" ::"i"(N) : "memory");
#endif
#endif
}

// ---------------- WMMA fragment helpers ------------------------------------
// A (16x32 bf16), ISA layout: lanes 0-15 hold row M=lane with K in
// {0..7, 16..23}; lanes 16-31 hold row M=lane-16 with K in {8..15, 24..31}.
__device__ __forceinline__ bf16x16 frag_a(const __bf16* __restrict__ base,
                                          size_t row, int ld, int k0, int lane) {
  const __bf16* p = base + row * (size_t)ld + k0 + 8 * (lane >> 4);
  bf16x8 lo = *(const bf16x8*)(p);
  bf16x8 hi = *(const bf16x8*)(p + 16);
  bf16x16 r;
#pragma unroll
  for (int i = 0; i < 8; i++) { r[i] = lo[i]; r[i + 8] = hi[i]; }
  return r;
}
// B (32x16 bf16): lane n<16 holds column n, K in {0..7, 16..23}; lane n+16
// holds column n, K in {8..15, 24..31}. `row` is the row of the K-major (B^T) source.
__device__ __forceinline__ bf16x16 frag_b(const __bf16* base,
                                          size_t row, int ld, int k0, int lane) {
  const __bf16* p = base + row * (size_t)ld + k0 + 8 * (lane >> 4);
  bf16x8 lo = *(const bf16x8*)(p);
  bf16x8 hi = *(const bf16x8*)(p + 16);
  bf16x16 r;
#pragma unroll
  for (int i = 0; i < 8; i++) { r[i] = lo[i]; r[i + 8] = hi[i]; }
  return r;
}
__device__ __forceinline__ f32x8 wmma_bf16(bf16x16 a, bf16x16 b, f32x8 c) {
  return __builtin_amdgcn_wmma_f32_16x16x32_bf16(false, a, false, b,
                                                 (short)0, c, false, false);
}

// ---------------- pack / prep kernels ---------------------------------------
__global__ void k_pack_x(const float* __restrict__ x, __bf16* __restrict__ xb) {
  int i = blockIdx.x * 256 + threadIdx.x;
  if (i < MROWS * CDIM) xb[i] = (__bf16)x[i];
}
__global__ void k_pack_w1(const float* __restrict__ wq, const float* __restrict__ wkv,
                          const float* __restrict__ wres, __bf16* __restrict__ w1) {
  int i = blockIdx.x * 256 + threadIdx.x;
  if (i >= 1024 * 256) return;
  int row = i >> 8, k = i & 255;
  float v = (row < 256) ? wq[i]
          : (row < 768) ? wkv[(size_t)(row - 256) * 256 + k]
                        : wres[(size_t)(row - 768) * 256 + k];
  w1[i] = (__bf16)v;
}
__global__ void k_pack_wp(const float* __restrict__ wproj, __bf16* __restrict__ wp) {
  int i = blockIdx.x * 256 + threadIdx.x;
  if (i < 256 * 256) wp[i] = (__bf16)wproj[i];
}
// insert -1 at the kernel center: full[c0][c1][t], t in [0,k*k)
__global__ void k_constrain(const float* __restrict__ kern, float* __restrict__ full,
                            int ksz) {
  int kk = ksz * ksz;
  int total = 64 * 64 * kk;
  int i = blockIdx.x * 256 + threadIdx.x;
  if (i >= total) return;
  int t = i % kk, cc = i / kk, ctr = kk / 2;
  float v = (t == ctr) ? -1.0f
          : (t < ctr) ? kern[(size_t)cc * (kk - 1) + t]
                      : kern[(size_t)cc * (kk - 1) + t - 1];
  full[i] = v;
}

// ---------------- fused projection GEMM: Y = x @ [wq;wkv;wres]^T ------------
__global__ __launch_bounds__(128) void k_proj_gemm(
    const __bf16* __restrict__ xb, const __bf16* __restrict__ w1,
    float* __restrict__ qf, __bf16* __restrict__ qb,
    __bf16* __restrict__ kb, __bf16* __restrict__ vt,
    float* __restrict__ resv) {
  int lane = threadIdx.x & 31, wave = threadIdx.x >> 5;
  int tile = blockIdx.x * 4 + wave;          // 0..36863
  int mt = tile >> 6, nt = tile & 63;
  int m0 = mt * 16, n0 = nt * 16;
  f32x8 acc = {};
#pragma unroll
  for (int kk = 0; kk < 256; kk += 32) {
    bf16x16 a = frag_a(xb, (size_t)(m0 + (lane & 15)), 256, kk, lane);
    bf16x16 b = frag_b(w1, (size_t)(n0 + (lane & 15)), 256, kk, lane);
    acc = wmma_bf16(a, b, acc);
  }
  int n = n0 + (lane & 15);
  int seg = n >> 8;                          // 0=q 1=k 2=v 3=res
  int c = n & 255;
  int mbase = m0 + 8 * (lane >> 4);
#pragma unroll
  for (int r = 0; r < 8; r++) {
    int m = mbase + r;
    int b_ = m / NTOK, nn = m % NTOK;        // tiles never straddle batches
    float val = acc[r];
    if (seg == 0) {
      int h = c >> 5, ch = c & 31;
      size_t idx = ((size_t)(b_ * 8 + h) * NTOK + nn) * 32 + ch;
      qf[idx] = val; qb[idx] = (__bf16)val;
    } else if (seg == 1) {
      int h = c >> 5, ch = c & 31;
      kb[((size_t)(b_ * 8 + h) * NTOK + nn) * 32 + ch] = (__bf16)val;
    } else if (seg == 2) {
      int h = c >> 5, ch = c & 31;
      vt[((size_t)(b_ * 8 + h) * 32 + ch) * NTOK + nn] = (__bf16)val;
    } else {
      resv[(size_t)(b_ * 256 + c) * NTOK + nn] = fmaxf(val, 0.0f);
    }
  }
}

// ---------------- flash attention -------------------------------------------
// grid (36, 32): blockIdx.y = b*8+h, 4 waves/block each own a 16-row tile.
// k/v 32x32 tiles staged in LDS (shared by all 4 waves), double-buffered via
// async global->LDS loads; s_wait_asynccnt + s_barrier for synchronization.
__global__ __launch_bounds__(128) void k_attn(
    const __bf16* __restrict__ qb, const __bf16* __restrict__ kb,
    const __bf16* __restrict__ vt, float* __restrict__ attnout) {
  __shared__ alignas(32) __bf16 kbuf[2][32][32];    // [buf][j][ch]
  __shared__ alignas(32) __bf16 vbuf[2][32][32];    // [buf][ch][j]
  __shared__ alignas(32) __bf16 stage[4][16][48];   // per-wave P 16x32 (+pad)
  int tid = threadIdx.x;
  int lane = tid & 31, wave = tid >> 5;
  int bh = blockIdx.y;
  int mt = blockIdx.x * 4 + wave;
  int m0 = mt * 16;
  const __bf16* qbase = qb + (size_t)bh * NTOK * 32;
  const __bf16* kbase = kb + (size_t)bh * NTOK * 32;
  const __bf16* vbase = vt + (size_t)bh * 32 * NTOK;
  const float scale = 0.17677669529663687f;         // 32^-0.5

  // per-thread 16B slice of a 32x32 tile: row sr, cols sc..sc+7
  int sr = tid >> 2;
  int sc = (tid & 3) * 8;

  bf16x16 aq = frag_a(qbase, (size_t)(m0 + (lane & 15)), 32, 0, lane);
  float mrow[8], lrow[8];
  f32x8 acc0 = {}, acc1 = {};
#pragma unroll
  for (int r = 0; r < 8; r++) { mrow[r] = -1e30f; lrow[r] = 0.0f; }

  // prologue: stage tile 0 into buffer 0 (2 async b128 per wave)
  stage16(kbase + (size_t)sr * 32 + sc,   &kbuf[0][sr][sc]);
  stage16(vbase + (size_t)sr * NTOK + sc, &vbuf[0][sr][sc]);

  for (int t = 0; t < NJT; ++t) {
    int cur = t & 1;
    if (t + 1 < NJT) {                      // prefetch next tile (uniform)
      int j0n = (t + 1) * 32;
      stage16(kbase + (size_t)(j0n + sr) * 32 + sc, &kbuf[cur ^ 1][sr][sc]);
      stage16(vbase + (size_t)sr * NTOK + j0n + sc, &vbuf[cur ^ 1][sr][sc]);
      wait_asynccnt<2>();                   // current tile's pair is done
    } else {
      wait_asynccnt<0>();
    }
    __syncthreads();

    const __bf16* kt = &kbuf[cur][0][0];
    const __bf16* vtile = &vbuf[cur][0][0];
    // scores: S = q @ k^T for a 16x32 column tile (two WMMAs, K=ch=32)
    bf16x16 bk0 = frag_b(kt, (size_t)(lane & 15), 32, 0, lane);
    bf16x16 bk1 = frag_b(kt, (size_t)(16 + (lane & 15)), 32, 0, lane);
    f32x8 z = {};
    f32x8 s0 = wmma_bf16(aq, bk0, z);
    f32x8 s1 = wmma_bf16(aq, bk1, z);
    float p0[8], p1[8];
#pragma unroll
    for (int r = 0; r < 8; r++) {
      // C-layout: lanes 0-15 hold row r (cols 0-15), lanes 16-31 row r+8.
      float a = s0[r] * scale, b = s1[r] * scale;
      float t0 = fmaxf(a, b);
      t0 = fmaxf(t0, __shfl_xor(t0, 1, 32));
      t0 = fmaxf(t0, __shfl_xor(t0, 2, 32));
      t0 = fmaxf(t0, __shfl_xor(t0, 4, 32));
      t0 = fmaxf(t0, __shfl_xor(t0, 8, 32));  // row max within 16-lane half
      float mnew = fmaxf(mrow[r], t0);
      float corr = __expf(mrow[r] - mnew);
      float e0 = __expf(a - mnew), e1 = __expf(b - mnew);
      float rs = e0 + e1;
      rs += __shfl_xor(rs, 1, 32);
      rs += __shfl_xor(rs, 2, 32);
      rs += __shfl_xor(rs, 4, 32);
      rs += __shfl_xor(rs, 8, 32);
      lrow[r] = lrow[r] * corr + rs;
      mrow[r] = mnew;
      acc0[r] *= corr; acc1[r] *= corr;
      p0[r] = e0; p1[r] = e1;
    }
    // C-layout -> A-layout via per-wave LDS staging
#pragma unroll
    for (int r = 0; r < 8; r++) {
      int ml = r + 8 * (lane >> 4);
      stage[wave][ml][(lane & 15)]      = (__bf16)p0[r];
      stage[wave][ml][16 + (lane & 15)] = (__bf16)p1[r];
    }
    __syncthreads();
    {
      const __bf16* pr = &stage[wave][lane & 15][0] + 8 * (lane >> 4);
      bf16x8 lo = *(const bf16x8*)(pr);
      bf16x8 hi = *(const bf16x8*)(pr + 16);
      bf16x16 pa;
#pragma unroll
      for (int i = 0; i < 8; i++) { pa[i] = lo[i]; pa[i + 8] = hi[i]; }
      // O += P @ V   (B columns = rows of the LDS Vt tile, contiguous K)
      bf16x16 bv0 = frag_b(vtile, (size_t)(lane & 15), 32, 0, lane);
      bf16x16 bv1 = frag_b(vtile, (size_t)(16 + (lane & 15)), 32, 0, lane);
      acc0 = wmma_bf16(pa, bv0, acc0);
      acc1 = wmma_bf16(pa, bv1, acc1);
    }
    __syncthreads();   // all waves done reading before next-tile DMA lands
  }
  int b_ = bh >> 3, h = bh & 7;
  int ch0 = lane & 15;
#pragma unroll
  for (int r = 0; r < 8; r++) {
    int m = m0 + r + 8 * (lane >> 4);
    float inv = 1.0f / lrow[r];
    size_t base = ((size_t)b_ * NTOK + m) * 256 + h * 32;
    attnout[base + ch0]      = acc0[r] * inv;
    attnout[base + 16 + ch0] = acc1[r] * inv;
  }
}

// ---------------- CRPE channel-mixing convs (chunks 1..3) -------------------
__global__ __launch_bounds__(256) void k_conv(
    const float* __restrict__ resv, const float* __restrict__ k3f,
    const float* __restrict__ k5f, const float* __restrict__ k7f,
    float* __restrict__ convbuf) {
  __shared__ float plane[NTOK];
  int tid = threadIdx.x;
  int b_ = blockIdx.x / 192;
  int co = blockIdx.x % 192;
  int chunk = co >> 6;                    // 0,1,2 -> k=3,5,7 on v chunks 1..3
  int ksz = 3 + 2 * chunk, pad = ksz >> 1;
  int cin_base = (chunk + 1) * 64;
  int coc = co & 63;
  const float* wptr = (chunk == 0) ? k3f : (chunk == 1) ? k5f : k7f;
  int py[9], px[9];
  float out[9];
#pragma unroll
  for (int i = 0; i < 9; i++) {
    int pix = tid + i * 256;
    py[i] = pix / 48; px[i] = pix % 48; out[i] = 0.0f;
  }
  for (int cin = 0; cin < 64; cin++) {
    __syncthreads();
    const float* src = resv + (size_t)(b_ * 256 + cin_base + cin) * NTOK;
#pragma unroll
    for (int i = 0; i < 9; i++) plane[tid + i * 256] = src[tid + i * 256];
    __syncthreads();
    if (cin + 1 < 64)                     // global_prefetch_b8 of next plane
      __builtin_prefetch(src + NTOK + (size_t)tid * 9, 0, 0);
    const float* wrow = wptr + (size_t)(coc * 64 + cin) * ksz * ksz;
    for (int ky = 0; ky < ksz; ky++) {
      for (int kx = 0; kx < ksz; kx++) {
        float w = wrow[ky * ksz + kx];
#pragma unroll
        for (int i = 0; i < 9; i++) {
          int yy = py[i] + ky - pad, xx = px[i] + kx - pad;
          float v = (yy >= 0 && yy < 48 && xx >= 0 && xx < 48)
                        ? plane[yy * 48 + xx] : 0.0f;
          out[i] += w * v;
        }
      }
    }
  }
  size_t obase = (size_t)(b_ * 192 + co) * NTOK;
#pragma unroll
  for (int i = 0; i < 9; i++) convbuf[obase + tid + i * 256] = out[i];
}

// ---------------- lowpass pooling: output grids s x s (s=1,2,3,6) ----------
__global__ void k_pool(const float* __restrict__ resv, float* __restrict__ pooled) {
  int idx = blockIdx.x * 256 + threadIdx.x;
  if (idx >= 12800) return;
  int s, chunk, rem;
  if (idx < 256)       { s = 1; chunk = 0; rem = idx; }
  else if (idx < 1280) { s = 2; chunk = 1; rem = idx - 256; }
  else if (idx < 3584) { s = 3; chunk = 2; rem = idx - 1280; }
  else                 { s = 6; chunk = 3; rem = idx - 3584; }
  int ss = s * s;
  int bc = rem / ss, cell = rem % ss;
  int b_ = bc >> 6, cc = bc & 63;
  int iy = cell / s, ix = cell % s;
  int win = 48 / s;
  const float* src = resv + (size_t)(b_ * 256 + chunk * 64 + cc) * NTOK;
  float sum = 0.0f;
  for (int yy = iy * win; yy < (iy + 1) * win; yy++)
    for (int xx = ix * win; xx < (ix + 1) * win; xx++)
      sum += src[yy * 48 + xx];
  pooled[idx] = sum / (float)(win * win);
}

// ---------------- combine: Z = attn + q*conv_v + relu(upsample(pool)) -------
__global__ void k_combine(
    const float* __restrict__ attnout, const float* __restrict__ qf,
    const float* __restrict__ resv, const float* __restrict__ convbuf,
    const float* __restrict__ pooled, __bf16* __restrict__ zb) {
  int idx = blockIdx.x * 256 + threadIdx.x;
  if (idx >= NB * NTOK * CDIM) return;
  int c = idx & 255;
  int nn = (idx >> 8) % NTOK;
  int b_ = idx / (NTOK * CDIM);
  int chunk = c >> 6, cc = c & 63;
  int h = c >> 5, ch = c & 31;
  float convv = (chunk == 0)
      ? resv[(size_t)(b_ * 256 + c) * NTOK + nn]
      : convbuf[(size_t)(b_ * 192 + (c - 64)) * NTOK + nn];
  float lp;
  if (chunk == 0) {
    lp = pooled[b_ * 64 + cc];                       // 1x1 -> broadcast
  } else {
    int s   = (chunk == 1) ? 2   : (chunk == 2) ? 3    : 6;
    int off = (chunk == 1) ? 256 : (chunk == 2) ? 1280 : 3584;
    const float* P = pooled + off + (size_t)(b_ * 64 + cc) * s * s;
    int y = nn / 48, x = nn % 48;
    float ys = y * (float)(s - 1) / 47.0f;
    float xs = x * (float)(s - 1) / 47.0f;
    int y0 = (int)floorf(ys), x0 = (int)floorf(xs);
    int y1 = (y0 + 1 < s - 1) ? y0 + 1 : s - 1;
    int x1 = (x0 + 1 < s - 1) ? x0 + 1 : s - 1;
    float wy = ys - (float)y0, wx = xs - (float)x0;
    float v00 = P[y0 * s + x0], v01 = P[y0 * s + x1];
    float v10 = P[y1 * s + x0], v11 = P[y1 * s + x1];
    lp = v00 * (1 - wy) * (1 - wx) + v01 * (1 - wy) * wx +
         v10 * wy * (1 - wx) + v11 * wy * wx;
  }
  lp = fmaxf(lp, 0.0f);
  float q = qf[((size_t)(b_ * 8 + h) * NTOK + nn) * 32 + ch];
  float a = attnout[((size_t)(b_ * NTOK + nn)) * 256 + c];
  zb[((size_t)(b_ * NTOK + nn)) * 256 + c] = (__bf16)(a + q * convv + lp);
}

// ---------------- output GEMM: out = Z @ wproj^T + bproj --------------------
__global__ __launch_bounds__(128) void k_out_gemm(
    const __bf16* __restrict__ zb, const __bf16* __restrict__ wp,
    const float* __restrict__ bproj, float* __restrict__ out) {
  int lane = threadIdx.x & 31, wave = threadIdx.x >> 5;
  int tile = blockIdx.x * 4 + wave;     // 0..9215
  int mt = tile >> 4, nt = tile & 15;
  int m0 = mt * 16, n0 = nt * 16;
  f32x8 acc = {};
#pragma unroll
  for (int kk = 0; kk < 256; kk += 32) {
    bf16x16 a = frag_a(zb, (size_t)(m0 + (lane & 15)), 256, kk, lane);
    bf16x16 b = frag_b(wp, (size_t)(n0 + (lane & 15)), 256, kk, lane);
    acc = wmma_bf16(a, b, acc);
  }
  int n = n0 + (lane & 15);
  float bias = bproj[n];
#pragma unroll
  for (int r = 0; r < 8; r++) {
    int m = m0 + r + 8 * (lane >> 4);
    out[(size_t)m * 256 + n] = acc[r] + bias;
  }
}

// ---------------------------------------------------------------------------
extern "C" void kernel_launch(void* const* d_in, const int* in_sizes, int n_in,
                              void* d_out, int out_size, void* d_ws, size_t ws_size,
                              hipStream_t stream) {
  const float* x     = (const float*)d_in[0];
  // d_in[1]=H, d_in[2]=W (compile-time constants 48x48)
  const float* wq    = (const float*)d_in[3];
  const float* wkv   = (const float*)d_in[4];
  const float* wres  = (const float*)d_in[5];
  const float* wproj = (const float*)d_in[6];
  const float* bproj = (const float*)d_in[7];
  const float* kern3 = (const float*)d_in[8];
  const float* kern5 = (const float*)d_in[9];
  const float* kern7 = (const float*)d_in[10];

  char* ws = (char*)d_ws;
  __bf16* xb   = (__bf16*)(ws + OFF_XB);
  __bf16* w1   = (__bf16*)(ws + OFF_W1);
  __bf16* wp   = (__bf16*)(ws + OFF_WP);
  float*  qf   = (float*) (ws + OFF_QF);
  __bf16* qbuf = (__bf16*)(ws + OFF_QB);
  __bf16* kbuf = (__bf16*)(ws + OFF_KB);
  __bf16* vt   = (__bf16*)(ws + OFF_VT);
  float*  resv = (float*) (ws + OFF_RES);
  float*  k3f  = (float*) (ws + OFF_K3);
  float*  k5f  = (float*) (ws + OFF_K5);
  float*  k7f  = (float*) (ws + OFF_K7);
  float*  cvb  = (float*) (ws + OFF_CV);
  float*  pool = (float*) (ws + OFF_PL);
  float*  att  = (float*) (ws + OFF_AT);
  __bf16* zb   = (__bf16*)(ws + OFF_ZB);
  float*  out  = (float*)d_out;

  // pack / prep
  k_pack_x  <<<MROWS * CDIM / 256, 256, 0, stream>>>(x, xb);
  k_pack_w1 <<<1024 * 256 / 256,   256, 0, stream>>>(wq, wkv, wres, w1);
  k_pack_wp <<<256 * 256 / 256,    256, 0, stream>>>(wproj, wp);
  k_constrain<<<(64 * 64 * 9  + 255) / 256, 256, 0, stream>>>(kern3, k3f, 3);
  k_constrain<<<(64 * 64 * 25 + 255) / 256, 256, 0, stream>>>(kern5, k5f, 5);
  k_constrain<<<(64 * 64 * 49 + 255) / 256, 256, 0, stream>>>(kern7, k7f, 7);

  // q/k/v/res projections (WMMA)
  k_proj_gemm<<<576 * 64 / 4, 128, 0, stream>>>(xb, w1, qf, qbuf, kbuf, vt, resv);

  // flash attention (WMMA + async LDS staging)
  k_attn<<<dim3(144 / 4, NB * NHEAD), 128, 0, stream>>>(qbuf, kbuf, vt, att);

  // CRPE convs + lowpass
  k_conv<<<NB * 192, 256, 0, stream>>>(resv, k3f, k5f, k7f, cvb);
  k_pool<<<(12800 + 255) / 256, 256, 0, stream>>>(resv, pool);
  k_combine<<<NB * NTOK * CDIM / 256, 256, 0, stream>>>(att, qf, resv, cvb, pool, zb);

  // final projection (WMMA)
  k_out_gemm<<<576 * 16 / 4, 128, 0, stream>>>(zb, wp, bproj, out);
}
